// BMN_66683662238004
// MI455X (gfx1250) — matrix operand order repaired and checked by reading, not compiled, not run on hardware
//
#include <hip/hip_runtime.h>
#include <hip/hip_bf16.h>
#include <cstdint>

// ---------------------------------------------------------------------------
// Types for CDNA5 WMMA (gfx1250, wave32)
// ---------------------------------------------------------------------------
typedef __attribute__((ext_vector_type(16))) __bf16 v16bf;
typedef __attribute__((ext_vector_type(2)))  __bf16 bf16x2;
typedef __attribute__((ext_vector_type(8)))  float  v8f;

union Frag {
    uint4  q[2];
    v16bf  v;
};

static __device__ __forceinline__ v8f wmma_bf16(const Frag& a, const Frag& b, v8f c) {
    // D = A(16x32 bf16) * B(32x16 bf16) + C(16x16 f32)
    return __builtin_amdgcn_wmma_f32_16x16x32_bf16(false, a.v, false, b.v,
                                                   (short)0, c, false, false);
}

// Native bf16 conversion: fptrunc pairs lower to v_cvt_pk_bf16_f32 on gfx1250.
static __device__ __forceinline__ uint32_t pack_bf16(float a, float b) {
    bf16x2 v;
    v[0] = (__bf16)a;
    v[1] = (__bf16)b;
    return __builtin_bit_cast(uint32_t, v);
}
static __device__ __forceinline__ uint16_t f2bf(float f) {
    return __builtin_bit_cast(uint16_t, (__bf16)f);
}
static __device__ __forceinline__ float bf2f(uint16_t h) {
    uint32_t u = ((uint32_t)h) << 16;
    return __builtin_bit_cast(float, u);
}
static __device__ __forceinline__ float sigmoidf(float x) {
    return 1.f / (1.f + __expf(-x));
}

// ---------------------------------------------------------------------------
// Problem constants
// ---------------------------------------------------------------------------
#define T_DIM   100
#define N_SMP   32
#define D_PROP  100
#define BATCH   4
#define FEAT    400
#define NP      10000          // D_PROP * T_DIM
#define NTILES  79             // ceil(10000 / 128)
#define P_PAD   (NTILES * 128) // 10112

// Workspace layout (bytes, all 256-aligned)
#define OFF_A5   ((size_t)0)          // w_c3d bf16 [512][4096]          4,194,304
#define OFF_A6   ((size_t)4194304)    // w_2d1 bf16 [128][512]             131,072
#define OFF_A7   ((size_t)4325376)    // w_2d2 bf16 [9][128][128]          294,912
#define OFF_B1   ((size_t)4620288)    // base1 f32 [4][256][100]           409,600
#define OFF_BASE ((size_t)5029888)    // base  f32 [4][128][100]           204,800
#define OFF_TEM  ((size_t)5234688)    // tem1  f32 [4][256][100]           409,600
#define OFF_LR   ((size_t)5644288)    // params lr  u32 [10112][32]      1,294,336
#define OFF_WL   ((size_t)6938624)    // params wl  f32                  1,294,336
#define OFF_WR   ((size_t)8232960)    // params wr  f32                  1,294,336
#define OFF_Y1   ((size_t)9527296)    // y1T bf16 [4][10000][512]       40,960,000
#define OFF_Y2   ((size_t)50487296)   // y2T bf16 [4][10000][128]       10,240,000
#define OFF_Y3   ((size_t)60727296)   // y3T bf16 [4][10000][128]       10,240,000

// ---------------------------------------------------------------------------
// Weight conversion kernels (f32 -> bf16)
// ---------------------------------------------------------------------------
__global__ void cvt_bf16_kernel(const float* __restrict__ src,
                                uint16_t* __restrict__ dst, int n) {
    int gid = blockIdx.x * 256 + threadIdx.x;
    if (gid < n) dst[gid] = f2bf(src[gid]);
}

// w_2d2 [o][c][kh][kw] -> A7 [s][o][c], s = kh*3+kw
__global__ void cvt_w2d2_kernel(const float* __restrict__ src,
                                uint16_t* __restrict__ dst) {
    int gid = blockIdx.x * 256 + threadIdx.x;
    if (gid >= 9 * 128 * 128) return;
    int s   = gid / 16384;
    int rem = gid % 16384;
    int o   = rem / 128;
    int c   = rem % 128;
    dst[gid] = f2bf(src[(o * 128 + c) * 9 + s]);
}

// ---------------------------------------------------------------------------
// BM sampling interpolation params: per (p = d*100+t, n): taps l,r + weights
// Faithful to the numpy construction (including xmax = j+1 as written).
// ---------------------------------------------------------------------------
__global__ void params_kernel(uint32_t* __restrict__ plr,
                              float* __restrict__ pwl,
                              float* __restrict__ pwr) {
    int gid = blockIdx.x * 256 + threadIdx.x;
    if (gid >= P_PAD * N_SMP) return;
    int p = gid >> 5;
    int n = gid & 31;
    uint32_t lrv = 0u;
    float wlv = 0.f, wrv = 0.f;
    if (p < NP) {
        int j = p / T_DIM;   // d (proposal length index)
        int i = p % T_DIM;   // t (start)
        if (i + j <= T_DIM - 2) {            // j < T-1-i (D=100 never binds)
            double length   = (double)(j + 1) - (double)i;   // as written
            double xmin_ext = (double)i - 0.5 * length;
            double bin      = (2.0 * length) / (double)(N_SMP - 1);
            double xp       = xmin_ext + (double)n * bin;
            if (xp >= 0.0 && xp <= (double)(T_DIM - 1)) {
                double l = floor(xp), r = ceil(xp);
                wlv = (float)(1.0 - (xp - l));
                wrv = (float)(1.0 - (r - xp));
                lrv = (uint32_t)(int)l | ((uint32_t)(int)r << 16);
            }
        }
    }
    plr[gid] = lrv; pwl[gid] = wlv; pwr[gid] = wrv;
}

// ---------------------------------------------------------------------------
// Conv1d (k=3, pad=1) + ReLU.  One block per (t, b); one thread per Cout.
// ---------------------------------------------------------------------------
__global__ void conv1d_relu_kernel(const float* __restrict__ in,
                                   const float* __restrict__ w,
                                   const float* __restrict__ bias,
                                   float* __restrict__ out,
                                   int Cin, int Cout) {
    __shared__ float col[1200];          // up to Cin=400, 3 taps
    const int t = blockIdx.x, b = blockIdx.y;
    const int n3 = Cin * 3;
    for (int i = threadIdx.x; i < n3; i += blockDim.x) {
        int kk = i / Cin, ci = i % Cin;
        int tt = t + kk - 1;
        col[kk * Cin + ci] =
            (tt >= 0 && tt < T_DIM) ? in[((size_t)b * Cin + ci) * T_DIM + tt] : 0.f;
    }
    __syncthreads();
    const int co = threadIdx.x;
    if (co < Cout) {
        float acc = bias[co];
        const float* wr = &w[(size_t)co * n3];   // [co][ci][k]
        for (int ci = 0; ci < Cin; ++ci) {
            acc += wr[ci * 3 + 0] * col[0 * Cin + ci]
                 + wr[ci * 3 + 1] * col[1 * Cin + ci]
                 + wr[ci * 3 + 2] * col[2 * Cin + ci];
        }
        out[((size_t)b * Cout + co) * T_DIM + t] = acc > 0.f ? acc : 0.f;
    }
}

// TEM head: conv1d 256->2 (k=3,pad=1) + sigmoid -> d_out[0..800)
__global__ void tem_out_kernel(const float* __restrict__ tem,
                               const float* __restrict__ w,
                               const float* __restrict__ bias,
                               float* __restrict__ out) {
    int gid = blockIdx.x * 256 + threadIdx.x;
    if (gid >= 800) return;
    int b = gid / 200, r = gid % 200, ch = r / 100, t = r % 100;
    float acc = bias[ch];
    for (int ci = 0; ci < 256; ++ci) {
        const float* wr = &w[(ch * 256 + ci) * 3];
        #pragma unroll
        for (int kk = 0; kk < 3; ++kk) {
            int tt = t + kk - 1;
            if (tt >= 0 && tt < T_DIM)
                acc += wr[kk] * tem[((size_t)b * 256 + ci) * T_DIM + tt];
        }
    }
    out[gid] = sigmoidf(acc);    // flat [b][ch][t] == b*200+ch*100+t
}

// ---------------------------------------------------------------------------
// Fused BM-sampling + Conv3d GEMM (the 167.8 GFLOP op).
//   y1[b,o,p] = relu( sum_{c,n} w_c3d[o,c,n] * (wl*base[b,c,l] + wr*base[b,c,r]) + bias[o] )
// GEMM: M=512 (o), N=10000 (p=d*100+t), K=4096 (c*32+n).  K-slab of 32 == one c.
// Workgroup: 256 thr (8 waves), 128x128 tile; wave = 32(M) x 64(N) via 8 WMMAs.
// base[b] lives in LDS; per-c the 32x128 bf16 B-slab is built in LDS.
// Output written transposed bf16: y1T[b][p][o] so the next GEMM reads K contiguously.
// ---------------------------------------------------------------------------
__global__ __launch_bounds__(256) void bm_gemm_kernel(
    const uint16_t* __restrict__ A5,   // bf16 [512][4096]
    const float*    __restrict__ base, // f32  [4][128][100]
    const float*    __restrict__ bias, // b_c3d [512]
    const uint32_t* __restrict__ plr,
    const float*    __restrict__ pwl,
    const float*    __restrict__ pwr,
    uint16_t*       __restrict__ y1T)  // bf16 [4][10000][512]
{
    __shared__ float    baseL[128 * 100];   // 51.2 KB
    __shared__ uint32_t GT[128 * 20];       // 10.24 KB: [p][n] bf16 pairs, padded rows

    const int tid  = threadIdx.x;
    const int lane = tid & 31, wave = tid >> 5;
    const int p0   = blockIdx.x * 128;
    const int o0   = blockIdx.y * 128;
    const int b    = blockIdx.z;

    for (int i = tid; i < 128 * 100; i += 256)
        baseL[i] = base[(size_t)b * 12800 + i];

    // Each thread owns 16 (p,n) sampling params: p = p0 + tid/2, n = (tid&1)*16 + q
    const int prow = tid >> 1;
    uint32_t lr[16]; float wl[16], wr[16];
    {
        size_t pb = ((size_t)(p0 + prow)) * 32 + (tid & 1) * 16;
        #pragma unroll
        for (int q = 0; q < 16; ++q) {
            lr[q] = plr[pb + q]; wl[q] = pwl[pb + q]; wr[q] = pwr[pb + q];
        }
    }
    __syncthreads();

    const int wm = wave >> 1, wn = wave & 1;
    const int khalf = lane >> 4, lrow = lane & 15;

    v8f acc[2][4];
    #pragma unroll
    for (int mi = 0; mi < 2; ++mi)
        #pragma unroll
        for (int ni = 0; ni < 4; ++ni)
            acc[mi][ni] = v8f{0.f,0.f,0.f,0.f,0.f,0.f,0.f,0.f};

    for (int c = 0; c < 128; ++c) {
        // ---- build bf16 B-slab G_T[p][n] for this channel ----
        const float* bc = &baseL[c * 100];
        uint32_t pk[8];
        #pragma unroll
        for (int q2 = 0; q2 < 8; ++q2) {
            uint32_t l0 = lr[2*q2] & 0xFFFFu,   r0 = lr[2*q2] >> 16;
            uint32_t l1 = lr[2*q2+1] & 0xFFFFu, r1 = lr[2*q2+1] >> 16;
            float g0 = wl[2*q2]   * bc[l0] + wr[2*q2]   * bc[r0];
            float g1 = wl[2*q2+1] * bc[l1] + wr[2*q2+1] * bc[r1];
            pk[q2] = pack_bf16(g0, g1);          // v_cvt_pk_bf16_f32
        }
        uint32_t* gdst = &GT[prow * 20 + (tid & 1) * 8];
        ((uint4*)gdst)[0] = make_uint4(pk[0], pk[1], pk[2], pk[3]);
        ((uint4*)gdst)[1] = make_uint4(pk[4], pk[5], pk[6], pk[7]);
        __syncthreads();

        // ---- WMMA over this K=32 slab ----
        Frag a[2];
        #pragma unroll
        for (int mi = 0; mi < 2; ++mi) {
            const uint4* ap = (const uint4*)(A5 +
                ((size_t)(o0 + wm * 32 + mi * 16 + lrow)) * 4096 + c * 32 + khalf * 16);
            a[mi].q[0] = ap[0];          // K = khalf*16 + 0..7
            a[mi].q[1] = ap[1];          // K = khalf*16 + 8..15
        }
        #pragma unroll
        for (int ni = 0; ni < 4; ++ni) {
            int col = wn * 64 + ni * 16 + lrow;
            const uint4* bp = (const uint4*)&GT[col * 20 + khalf * 8];
            Frag bb;
            bb.q[0] = bp[0]; bb.q[1] = bp[1];    // K = khalf*16 + 0..15
            acc[0][ni] = wmma_bf16(a[0], bb, acc[0][ni]);
            acc[1][ni] = wmma_bf16(a[1], bb, acc[1][ni]);
        }
        __syncthreads();
    }

    // ---- epilogue: bias + relu, store bf16 transposed [b][p][o] ----
    #pragma unroll
    for (int mi = 0; mi < 2; ++mi) {
        int ob = o0 + wm * 32 + mi * 16 + khalf * 8;   // 8 contiguous o per lane
        float bs[8];
        #pragma unroll
        for (int r = 0; r < 8; ++r) bs[r] = bias[ob + r];
        #pragma unroll
        for (int ni = 0; ni < 4; ++ni) {
            int p = p0 + wn * 64 + ni * 16 + lrow;
            if (p < NP) {
                uint32_t pk2[4];
                #pragma unroll
                for (int r2 = 0; r2 < 4; ++r2) {
                    float v0 = acc[mi][ni][2*r2]   + bs[2*r2];
                    float v1 = acc[mi][ni][2*r2+1] + bs[2*r2+1];
                    v0 = v0 > 0.f ? v0 : 0.f;
                    v1 = v1 > 0.f ? v1 : 0.f;
                    pk2[r2] = pack_bf16(v0, v1);
                }
                *(uint4*)&y1T[((size_t)b * NP + p) * 512 + ob] =
                    make_uint4(pk2[0], pk2[1], pk2[2], pk2[3]);
            }
        }
    }
}

// ---------------------------------------------------------------------------
// 1x1 conv as streaming WMMA GEMM: M=128, N tile=128, K runtime (512).
// B-fragments are contiguous 128-bit loads from the transposed activation.
// ---------------------------------------------------------------------------
__global__ __launch_bounds__(256) void gemm1x1_kernel(
    const uint16_t* __restrict__ A,   // bf16 [128][K]
    const uint16_t* __restrict__ xT,  // bf16 [4][10000][K]
    const float*    __restrict__ bias,
    uint16_t*       __restrict__ yT,  // bf16 [4][10000][128]
    int K)
{
    const int tid  = threadIdx.x;
    const int lane = tid & 31, wave = tid >> 5;
    const int p0   = blockIdx.x * 128;
    const int b    = blockIdx.y;
    const int wm = wave >> 1, wn = wave & 1;
    const int khalf = lane >> 4, lrow = lane & 15;

    v8f acc[2][4];
    #pragma unroll
    for (int mi = 0; mi < 2; ++mi)
        #pragma unroll
        for (int ni = 0; ni < 4; ++ni)
            acc[mi][ni] = v8f{0.f,0.f,0.f,0.f,0.f,0.f,0.f,0.f};

    int pc[4];
    #pragma unroll
    for (int ni = 0; ni < 4; ++ni) {
        int p = p0 + wn * 64 + ni * 16 + lrow;
        pc[ni] = p < NP ? p : NP - 1;    // clamp; stores are guarded
    }

    const int ksteps = K >> 5;
    for (int kk = 0; kk < ksteps; ++kk) {
        const int k0 = kk * 32;
        Frag a[2];
        #pragma unroll
        for (int mi = 0; mi < 2; ++mi) {
            const uint4* ap = (const uint4*)(A +
                (size_t)(wm * 32 + mi * 16 + lrow) * K + k0 + khalf * 16);
            a[mi].q[0] = ap[0]; a[mi].q[1] = ap[1];
        }
        #pragma unroll
        for (int ni = 0; ni < 4; ++ni) {
            const uint4* bp = (const uint4*)(xT +
                ((size_t)b * NP + pc[ni]) * K + k0 + khalf * 16);
            Frag bb;
            bb.q[0] = bp[0]; bb.q[1] = bp[1];
            acc[0][ni] = wmma_bf16(a[0], bb, acc[0][ni]);
            acc[1][ni] = wmma_bf16(a[1], bb, acc[1][ni]);
        }
    }

    #pragma unroll
    for (int mi = 0; mi < 2; ++mi) {
        int ob = wm * 32 + mi * 16 + khalf * 8;
        float bs[8];
        #pragma unroll
        for (int r = 0; r < 8; ++r) bs[r] = bias[ob + r];
        #pragma unroll
        for (int ni = 0; ni < 4; ++ni) {
            int p = p0 + wn * 64 + ni * 16 + lrow;
            if (p < NP) {
                uint32_t pk2[4];
                #pragma unroll
                for (int r2 = 0; r2 < 4; ++r2) {
                    float v0 = acc[mi][ni][2*r2]   + bs[2*r2];
                    float v1 = acc[mi][ni][2*r2+1] + bs[2*r2+1];
                    v0 = v0 > 0.f ? v0 : 0.f;
                    v1 = v1 > 0.f ? v1 : 0.f;
                    pk2[r2] = pack_bf16(v0, v1);
                }
                *(uint4*)&yT[((size_t)b * NP + p) * 128 + ob] =
                    make_uint4(pk2[0], pk2[1], pk2[2], pk2[3]);
            }
        }
    }
}

// ---------------------------------------------------------------------------
// 3x3 conv (pad=1) on the 100x100 (d,t) image as 9 shifted WMMA GEMMs.
// K loop: 9 shifts x 4 chunks of 32 channels. Out-of-image taps -> zero frags.
// ---------------------------------------------------------------------------
__global__ __launch_bounds__(256) void conv3x3_kernel(
    const uint16_t* __restrict__ A,   // bf16 [9][128][128]  (A[s][o][c])
    const uint16_t* __restrict__ xT,  // bf16 [4][10000][128]
    const float*    __restrict__ bias,
    uint16_t*       __restrict__ yT)  // bf16 [4][10000][128]
{
    const int tid  = threadIdx.x;
    const int lane = tid & 31, wave = tid >> 5;
    const int p0   = blockIdx.x * 128;
    const int b    = blockIdx.y;
    const int wm = wave >> 1, wn = wave & 1;
    const int khalf = lane >> 4, lrow = lane & 15;

    v8f acc[2][4];
    #pragma unroll
    for (int mi = 0; mi < 2; ++mi)
        #pragma unroll
        for (int ni = 0; ni < 4; ++ni)
            acc[mi][ni] = v8f{0.f,0.f,0.f,0.f,0.f,0.f,0.f,0.f};

    int pn[4], dn[4], tn[4];
    #pragma unroll
    for (int ni = 0; ni < 4; ++ni) {
        int p = p0 + wn * 64 + ni * 16 + lrow;
        pn[ni] = p; dn[ni] = p / 100; tn[ni] = p % 100;
    }

    for (int s = 0; s < 9; ++s) {
        const int dy = s / 3 - 1, dx = s % 3 - 1;
        int  pofs[4]; bool ok[4];
        #pragma unroll
        for (int ni = 0; ni < 4; ++ni) {
            int dd = dn[ni] + dy, tt = tn[ni] + dx;
            ok[ni]   = (pn[ni] < NP) && dd >= 0 && dd < 100 && tt >= 0 && tt < 100;
            pofs[ni] = dd * 100 + tt;
        }
        for (int cb = 0; cb < 4; ++cb) {
            Frag a[2];
            #pragma unroll
            for (int mi = 0; mi < 2; ++mi) {
                const uint4* ap = (const uint4*)(A +
                    ((size_t)(s * 128 + wm * 32 + mi * 16 + lrow)) * 128 +
                    cb * 32 + khalf * 16);
                a[mi].q[0] = ap[0]; a[mi].q[1] = ap[1];
            }
            #pragma unroll
            for (int ni = 0; ni < 4; ++ni) {
                Frag bb;
                if (ok[ni]) {
                    const uint4* bp = (const uint4*)(xT +
                        ((size_t)b * NP + pofs[ni]) * 128 + cb * 32 + khalf * 16);
                    bb.q[0] = bp[0]; bb.q[1] = bp[1];
                } else {
                    bb.q[0] = make_uint4(0u,0u,0u,0u);
                    bb.q[1] = make_uint4(0u,0u,0u,0u);
                }
                acc[0][ni] = wmma_bf16(a[0], bb, acc[0][ni]);
                acc[1][ni] = wmma_bf16(a[1], bb, acc[1][ni]);
            }
        }
    }

    #pragma unroll
    for (int mi = 0; mi < 2; ++mi) {
        int ob = wm * 32 + mi * 16 + khalf * 8;
        float bs[8];
        #pragma unroll
        for (int r = 0; r < 8; ++r) bs[r] = bias[ob + r];
        #pragma unroll
        for (int ni = 0; ni < 4; ++ni) {
            int p = p0 + wn * 64 + ni * 16 + lrow;
            if (p < NP) {
                uint32_t pk2[4];
                #pragma unroll
                for (int r2 = 0; r2 < 4; ++r2) {
                    float v0 = acc[mi][ni][2*r2]   + bs[2*r2];
                    float v1 = acc[mi][ni][2*r2+1] + bs[2*r2+1];
                    v0 = v0 > 0.f ? v0 : 0.f;
                    v1 = v1 > 0.f ? v1 : 0.f;
                    pk2[r2] = pack_bf16(v0, v1);
                }
                *(uint4*)&yT[((size_t)b * NP + p) * 128 + ob] =
                    make_uint4(pk2[0], pk2[1], pk2[2], pk2[3]);
            }
        }
    }
}

// ---------------------------------------------------------------------------
// Final 1x1 conv 128->2 + sigmoid -> d_out[800 ..)
// ---------------------------------------------------------------------------
__global__ void final_kernel(const uint16_t* __restrict__ y3T,
                             const float* __restrict__ w3,   // [2][128]
                             const float* __restrict__ b3,
                             float* __restrict__ out) {
    int gid = blockIdx.x * 256 + threadIdx.x;
    if (gid >= BATCH * NP) return;
    int b = gid / NP, p = gid % NP;
    const uint16_t* row = y3T + ((size_t)b * NP + p) * 128;
    float a0 = b3[0], a1 = b3[1];
    for (int c = 0; c < 128; ++c) {
        float v = bf2f(row[c]);
        a0 += v * w3[c];
        a1 += v * w3[128 + c];
    }
    out[800 + ((size_t)b * 2 + 0) * NP + p] = sigmoidf(a0);
    out[800 + ((size_t)b * 2 + 1) * NP + p] = sigmoidf(a1);
}

// ---------------------------------------------------------------------------
// Launch
// ---------------------------------------------------------------------------
extern "C" void kernel_launch(void* const* d_in, const int* in_sizes, int n_in,
                              void* d_out, int out_size, void* d_ws, size_t ws_size,
                              hipStream_t stream) {
    const float* x      = (const float*)d_in[0];
    const float* w_b1   = (const float*)d_in[1];
    const float* b_b1   = (const float*)d_in[2];
    const float* w_b2   = (const float*)d_in[3];
    const float* b_b2   = (const float*)d_in[4];
    const float* w_t1   = (const float*)d_in[5];
    const float* b_t1   = (const float*)d_in[6];
    const float* w_t2   = (const float*)d_in[7];
    const float* b_t2   = (const float*)d_in[8];
    const float* w_c3d  = (const float*)d_in[9];
    const float* b_c3d  = (const float*)d_in[10];
    const float* w_2d1  = (const float*)d_in[11];
    const float* b_2d1  = (const float*)d_in[12];
    const float* w_2d2  = (const float*)d_in[13];
    const float* b_2d2  = (const float*)d_in[14];
    const float* w_2d3  = (const float*)d_in[15];
    const float* b_2d3  = (const float*)d_in[16];

    uint8_t* ws = (uint8_t*)d_ws;
    uint16_t* A5   = (uint16_t*)(ws + OFF_A5);
    uint16_t* A6   = (uint16_t*)(ws + OFF_A6);
    uint16_t* A7   = (uint16_t*)(ws + OFF_A7);
    float*    bs1  = (float*)   (ws + OFF_B1);
    float*    base = (float*)   (ws + OFF_BASE);
    float*    tem1 = (float*)   (ws + OFF_TEM);
    uint32_t* plr  = (uint32_t*)(ws + OFF_LR);
    float*    pwl  = (float*)   (ws + OFF_WL);
    float*    pwr  = (float*)   (ws + OFF_WR);
    uint16_t* y1T  = (uint16_t*)(ws + OFF_Y1);
    uint16_t* y2T  = (uint16_t*)(ws + OFF_Y2);
    uint16_t* y3T  = (uint16_t*)(ws + OFF_Y3);
    float*    out  = (float*)d_out;

    // weight conversions + sampling params (independent of activations)
    cvt_bf16_kernel<<<(512*4096+255)/256, 256, 0, stream>>>(w_c3d, A5, 512*4096);
    cvt_bf16_kernel<<<(128*512+255)/256,  256, 0, stream>>>(w_2d1, A6, 128*512);
    cvt_w2d2_kernel<<<(9*128*128+255)/256,256, 0, stream>>>(w_2d2, A7);
    params_kernel<<<(P_PAD*N_SMP+255)/256,256, 0, stream>>>(plr, pwl, pwr);

    // base + TEM branch (small conv1d stack)
    conv1d_relu_kernel<<<dim3(T_DIM, BATCH), 256, 0, stream>>>(x,    w_b1, b_b1, bs1, 400, 256);
    conv1d_relu_kernel<<<dim3(T_DIM, BATCH), 128, 0, stream>>>(bs1,  w_b2, b_b2, base, 256, 128);
    conv1d_relu_kernel<<<dim3(T_DIM, BATCH), 256, 0, stream>>>(base, w_t1, b_t1, tem1, 128, 256);
    tem_out_kernel<<<4, 256, 0, stream>>>(tem1, w_t2, b_t2, out);

    // fused BM-sampling + Conv3d GEMM (dominant op), then the 2D head
    bm_gemm_kernel<<<dim3(NTILES, 4, BATCH), 256, 0, stream>>>(A5, base, b_c3d,
                                                               plr, pwl, pwr, y1T);
    gemm1x1_kernel<<<dim3(NTILES, BATCH), 256, 0, stream>>>(A6, y1T, b_2d1, y2T, 512);
    conv3x3_kernel<<<dim3(NTILES, BATCH), 256, 0, stream>>>(A7, y2T, b_2d2, y3T);
    final_kernel<<<(BATCH*NP+255)/256, 256, 0, stream>>>(y3T, w_2d3, b_2d3, out);
}
